// AttentionLayerDecoder_75204877353713
// MI455X (gfx1250) — matrix operand order repaired and checked by reading, not compiled, not run on hardware
//
#include <hip/hip_runtime.h>
#include <hip/hip_bf16.h>

// ---------------- problem constants (fixed by the reference) ----------------
constexpr int Bb   = 2;
constexpr int Nn   = 2048;
constexpr int HD   = 2048;
constexpr int NH   = 16;
constexpr int KVH  = 4;
constexpr int Dd   = HD / NH;     // 128
constexpr int KVD  = KVH * Dd;    // 512
constexpr int ROWS = Bb * Nn;     // 4096

typedef unsigned short u16;
typedef __attribute__((ext_vector_type(16))) __bf16 v16bf;
typedef __attribute__((ext_vector_type(8)))  float  v8f;

union FragBF {
  v16bf v;
  u16   u[16];
  uint4 q[2];
};

static __device__ __forceinline__ u16 f32_to_bf16(float f) {
  union { float f; unsigned u; } x; x.f = f;
  unsigned r = x.u + 0x7FFFu + ((x.u >> 16) & 1u);   // round-to-nearest-even
  return (u16)(r >> 16);
}

static __device__ __forceinline__ v8f wmma_bf16(v16bf a, v16bf b, v8f c) {
  // D(16x16,f32) = A(16x32,bf16) x B(32x16,bf16) + C
  return __builtin_amdgcn_wmma_f32_16x16x32_bf16(false, a, false, b, (short)0, c,
                                                 false, false);
}

// CDNA5 async memory->LDS copy (GLOBAL_LOAD_ASYNC_TO_LDS_B128, ASYNCcnt).
// Generic pointers to __shared__ carry the LDS byte offset in their low 32
// bits (aperture mapping truncates to addr[31:0]), which is exactly the
// per-lane LDS address VGPR the instruction expects.
static __device__ __forceinline__ void async_g2lds_b128(void* lds, const void* gp) {
  unsigned l = (unsigned)(size_t)lds;
  asm volatile("global_load_async_to_lds_b128 %0, %1, off"
               :: "v"(l), "v"(gp) : "memory");
}
static __device__ __forceinline__ void wait_async0() {
  asm volatile("s_wait_asynccnt 0" ::: "memory");
}

// ---------------------------------------------------------------------------
// Kernel 1: RMSNorm (eps = FLT_EPSILON) * weight, output bf16
// ---------------------------------------------------------------------------
__global__ __launch_bounds__(256) void rmsnorm_bf16_kernel(
    const float* __restrict__ x, const float* __restrict__ w, u16* __restrict__ xh) {
  const int row  = blockIdx.x;
  const int lane = threadIdx.x & 31;
  const int wave = threadIdx.x >> 5;
  const float* xr = x + (size_t)row * HD;

  float ss = 0.f;
  #pragma unroll
  for (int c = threadIdx.x; c < HD; c += 256) { float v = xr[c]; ss = fmaf(v, v, ss); }
  ss += __shfl_xor(ss, 1, 32);  ss += __shfl_xor(ss, 2, 32);
  ss += __shfl_xor(ss, 4, 32);  ss += __shfl_xor(ss, 8, 32);
  ss += __shfl_xor(ss, 16, 32);
  __shared__ float red[8];
  if (lane == 0) red[wave] = ss;
  __syncthreads();
  float tot = red[0] + red[1] + red[2] + red[3] + red[4] + red[5] + red[6] + red[7];
  const float inv = rsqrtf(tot * (1.0f / (float)HD) + 1.1920928955078125e-07f);
  #pragma unroll
  for (int c = threadIdx.x; c < HD; c += 256)
    xh[(size_t)row * HD + c] = f32_to_bf16(xr[c] * inv * w[c]);
}

// ---------------------------------------------------------------------------
// Kernel 2: f32 -> bf16 conversion (weights)
// ---------------------------------------------------------------------------
__global__ __launch_bounds__(256) void cvt_f32_bf16_kernel(
    const float* __restrict__ in, u16* __restrict__ out, int n) {
  int i = blockIdx.x * 256 + threadIdx.x;
  if (i < n) out[i] = f32_to_bf16(in[i]);
}

// ---------------------------------------------------------------------------
// Kernel 3: C[M,Nc] = A[M,K](bf16) * W[Nc,K]^T(bf16) + bias, out bf16 or f32.
// 128x128 block tile, BK=32, 8 waves each computing 64x32 (4x2 WMMA tiles).
// Double-buffered LDS fed by async global->LDS copies; padded rows to kill
// bank conflicts on fragment reads.
// ---------------------------------------------------------------------------
#define BM 128
#define BN 128
#define BK 32
#define LKP (BK + 8)   // padded row: 40 halves = 80B (16B-aligned, conflict-free)

__global__ __launch_bounds__(256) void gemm_bf16_wmma_kernel(
    const u16* __restrict__ A, const u16* __restrict__ W,
    const float* __restrict__ bias, void* __restrict__ outp,
    int M, int Ncols, int K, int out_f32) {
  __shared__ __align__(16) u16 sA[2][BM][LKP];
  __shared__ __align__(16) u16 sB[2][BN][LKP];

  const int tid  = threadIdx.x;
  const int wave = tid >> 5;
  const int lane = tid & 31;
  const int lrow = lane & 15;
  const int lhi  = lane >> 4;        // 0/1
  const int mw   = wave >> 2;        // 0..1
  const int nw   = wave & 3;         // 0..3
  const int mBase = blockIdx.y * BM;
  const int nBase = blockIdx.x * BN;

  v8f acc[4][2] = {};

  const int ldRow  = tid >> 1;           // 0..127
  const int ldHalf = (tid & 1) * 16;     // 0 or 16 (in halves)
  const u16* gA = A + (size_t)(mBase + ldRow) * K + ldHalf;
  const u16* gB = W + (size_t)(nBase + ldRow) * K + ldHalf;

  auto stage = [&](int buf, int kb) {
    async_g2lds_b128(&sA[buf][ldRow][ldHalf],     gA + kb);
    async_g2lds_b128(&sA[buf][ldRow][ldHalf + 8], gA + kb + 8);
    async_g2lds_b128(&sB[buf][ldRow][ldHalf],     gB + kb);
    async_g2lds_b128(&sB[buf][ldRow][ldHalf + 8], gB + kb + 8);
  };

  stage(0, 0);
  wait_async0();
  __syncthreads();

  int buf = 0;
  for (int kb = 0; kb < K; kb += BK) {
    if (kb + BK < K) stage(buf ^ 1, kb + BK);   // overlap with compute below

    FragBF bf[2];
    #pragma unroll
    for (int ns = 0; ns < 2; ++ns) {
      const u16* p = &sB[buf][nw * 32 + ns * 16 + lrow][lhi * 16];
      bf[ns].q[0] = ((const uint4*)p)[0];
      bf[ns].q[1] = ((const uint4*)p)[1];
    }
    FragBF af[4];
    #pragma unroll
    for (int ms = 0; ms < 4; ++ms) {
      const u16* p = &sA[buf][mw * 64 + ms * 16 + lrow][0];
      af[ms].q[0] = *(const uint4*)(p + lhi * 16);
      af[ms].q[1] = *(const uint4*)(p + lhi * 16 + 8);
    }
    #pragma unroll
    for (int ms = 0; ms < 4; ++ms)
      #pragma unroll
      for (int ns = 0; ns < 2; ++ns)
        acc[ms][ns] = wmma_bf16(af[ms].v, bf[ns].v, acc[ms][ns]);

    wait_async0();
    __syncthreads();
    buf ^= 1;
  }

  // epilogue: C layout — col = lrow (per lane), row = r + 8*lhi (per VGPR)
  #pragma unroll
  for (int ms = 0; ms < 4; ++ms) {
    #pragma unroll
    for (int ns = 0; ns < 2; ++ns) {
      const int col = nBase + nw * 32 + ns * 16 + lrow;
      const float bv = bias[col];
      #pragma unroll
      for (int r = 0; r < 8; ++r) {
        const int rowi = mBase + mw * 64 + ms * 16 + r + 8 * lhi;
        const float v = acc[ms][ns][r] + bv;
        if (out_f32) ((float*)outp)[(size_t)rowi * Ncols + col] = v;
        else         ((u16*)outp)[(size_t)rowi * Ncols + col]   = f32_to_bf16(v);
      }
    }
  }
}

// ---------------------------------------------------------------------------
// Kernel 4: flash attention, inverted causal mask (j <= i masked to -1e9),
// NO 1/sqrt(D) scaling, GQA (head h uses kv head h % KVH).
// One wave per 16-query tile; block = 8 waves = 128 query rows, all sharing
// the same kv head -> K and V^T tiles staged cooperatively per block.
// ---------------------------------------------------------------------------
__global__ __launch_bounds__(256) void attn_wmma_kernel(
    const u16* __restrict__ Qh, const u16* __restrict__ Kh,
    const u16* __restrict__ Vh, u16* __restrict__ Ah) {
  const int tid  = threadIdx.x;
  const int wave = tid >> 5;
  const int lane = tid & 31;
  const int lrow = lane & 15;
  const int lhi  = lane >> 4;

  const int nRowBlks = Nn / 128;                 // 16
  const int bh = blockIdx.x / nRowBlks;          // 0..31
  const int rb = blockIdx.x % nRowBlks;
  const int b  = bh / NH;
  const int h  = bh % NH;
  const int g  = h % KVH;
  const int itile = rb * 128 + wave * 16;        // first query row of this wave

  __shared__ __align__(16) u16 sK[32][136];      // K tile, row-major, padded
  __shared__ __align__(16) u16 sVT[128][40];     // V tile transposed, padded
  __shared__ __align__(16) u16 sP[8][16][40];    // per-wave P staging, padded

  // Q fragments (16 rows x 128 d), 4 K-chunks of 32
  FragBF qf[4];
  {
    const u16* qbase = Qh + ((size_t)(b * Nn) + itile) * HD + h * Dd;
    #pragma unroll
    for (int c = 0; c < 4; ++c) {
      const u16* p = qbase + (size_t)lrow * HD + c * 32;
      qf[c].q[0] = *(const uint4*)(p + lhi * 16);
      qf[c].q[1] = *(const uint4*)(p + lhi * 16 + 8);
    }
  }

  v8f   o[8] = {};                               // 16 x 128 f32 output accum
  float mrow[8], lsum[8];
  #pragma unroll
  for (int r = 0; r < 8; ++r) { mrow[r] = -3.0e38f; lsum[r] = 0.f; }

  const u16* kbase = Kh + (size_t)(b * Nn) * KVD + g * Dd;
  const u16* vbase = Vh + (size_t)(b * Nn) * KVD + g * Dd;

  const int krow = tid >> 3;          // 0..31  (staging row)
  const int kcol = (tid & 7) * 16;    // 0..112 (staging column, halves)

  for (int j0 = 0; j0 < Nn; j0 += 32) {
    // ---- cooperative staging of K tile (async) and V^T tile ----
    {
      const u16* kg = kbase + (size_t)(j0 + krow) * KVD + kcol;
      async_g2lds_b128(&sK[krow][kcol],     kg);
      async_g2lds_b128(&sK[krow][kcol + 8], kg + 8);
      const u16* vg = vbase + (size_t)(j0 + krow) * KVD + kcol;
      FragBF t;
      t.q[0] = *(const uint4*)vg;
      t.q[1] = *(const uint4*)(vg + 8);
      #pragma unroll
      for (int e = 0; e < 16; ++e) sVT[kcol + e][krow] = t.u[e];
      wait_async0();
    }
    __syncthreads();

    // ---- S = Q K^T for 32 keys (two 16-key tiles) ----
    v8f s0 = {}, s1 = {};
    #pragma unroll
    for (int c = 0; c < 4; ++c) {
      FragBF k0, k1;
      const u16* p0 = &sK[lrow][c * 32 + lhi * 16];
      k0.q[0] = ((const uint4*)p0)[0];
      k0.q[1] = ((const uint4*)p0)[1];
      const u16* p1 = &sK[16 + lrow][c * 32 + lhi * 16];
      k1.q[0] = ((const uint4*)p1)[0];
      k1.q[1] = ((const uint4*)p1)[1];
      s0 = wmma_bf16(qf[c].v, k0.v, s0);
      s1 = wmma_bf16(qf[c].v, k1.v, s1);
    }

    // ---- masked online softmax update ----
    #pragma unroll
    for (int r = 0; r < 8; ++r) {
      const int irow = itile + r + 8 * lhi;
      float a0 = (j0 + lrow      <= irow) ? -1e9f : s0[r];
      float a1 = (j0 + 16 + lrow <= irow) ? -1e9f : s1[r];
      float mt = fmaxf(a0, a1);
      mt = fmaxf(mt, __shfl_xor(mt, 1, 32));
      mt = fmaxf(mt, __shfl_xor(mt, 2, 32));
      mt = fmaxf(mt, __shfl_xor(mt, 4, 32));
      mt = fmaxf(mt, __shfl_xor(mt, 8, 32));
      const float mnew  = fmaxf(mrow[r], mt);
      const float alpha = __expf(mrow[r] - mnew);
      const float p0 = __expf(a0 - mnew);
      const float p1 = __expf(a1 - mnew);
      float ps = p0 + p1;
      ps += __shfl_xor(ps, 1, 32);
      ps += __shfl_xor(ps, 2, 32);
      ps += __shfl_xor(ps, 4, 32);
      ps += __shfl_xor(ps, 8, 32);
      lsum[r] = lsum[r] * alpha + ps;
      mrow[r] = mnew;
      #pragma unroll
      for (int c = 0; c < 8; ++c) o[c][r] = o[c][r] * alpha;
      sP[wave][r + 8 * lhi][lrow]      = f32_to_bf16(p0);
      sP[wave][r + 8 * lhi][16 + lrow] = f32_to_bf16(p1);
    }
    __syncthreads();

    // ---- O += P(16x32) @ V(32x128), V^T fragments straight from LDS ----
    FragBF pf;
    pf.q[0] = *(const uint4*)&sP[wave][lrow][lhi * 16];
    pf.q[1] = *(const uint4*)&sP[wave][lrow][lhi * 16 + 8];
    #pragma unroll
    for (int c = 0; c < 8; ++c) {
      FragBF vf;
      const u16* p = &sVT[c * 16 + lrow][lhi * 16];
      vf.q[0] = ((const uint4*)p)[0];
      vf.q[1] = ((const uint4*)p)[1];
      o[c] = wmma_bf16(pf.v, vf.v, o[c]);
    }
    __syncthreads();   // protect sK/sVT/sP before next iteration's staging
  }

  // ---- normalize and store (bf16) ----
  u16* obase = Ah + ((size_t)(b * Nn) + itile) * HD + h * Dd;
  #pragma unroll
  for (int c = 0; c < 8; ++c) {
    #pragma unroll
    for (int r = 0; r < 8; ++r) {
      const float val = o[c][r] / lsum[r];
      obase[(size_t)(r + 8 * lhi) * HD + c * 16 + lrow] = f32_to_bf16(val);
    }
  }
}

// ---------------------------------------------------------------------------
// host-side orchestration
// ---------------------------------------------------------------------------
extern "C" void kernel_launch(void* const* d_in, const int* in_sizes, int n_in,
                              void* d_out, int out_size, void* d_ws, size_t ws_size,
                              hipStream_t stream) {
  (void)in_sizes; (void)n_in; (void)out_size; (void)ws_size;

  const float* tokens = (const float*)d_in[0];
  const float* norm_w = (const float*)d_in[1];
  const float* Wq = (const float*)d_in[2];
  const float* bq = (const float*)d_in[3];
  const float* Wk = (const float*)d_in[4];
  const float* bk = (const float*)d_in[5];
  const float* Wv = (const float*)d_in[6];
  const float* bv = (const float*)d_in[7];
  const float* Wo = (const float*)d_in[8];
  const float* bo = (const float*)d_in[9];
  float* out = (float*)d_out;

  // workspace carve-up (bf16 buffers), 256B aligned
  char* ws = (char*)d_ws;
  size_t off = 0;
  auto alloc = [&](size_t bytes) -> u16* {
    off = (off + 255) & ~(size_t)255;
    u16* p = (u16*)(ws + off);
    off += bytes;
    return p;
  };
  u16* xh   = alloc((size_t)ROWS * HD * 2);
  u16* Wq_h = alloc((size_t)HD * HD * 2);
  u16* Wk_h = alloc((size_t)KVD * HD * 2);
  u16* Wv_h = alloc((size_t)KVD * HD * 2);
  u16* Wo_h = alloc((size_t)HD * HD * 2);
  u16* Qbuf = alloc((size_t)ROWS * HD * 2);
  u16* Kbuf = alloc((size_t)ROWS * KVD * 2);
  u16* Vbuf = alloc((size_t)ROWS * KVD * 2);
  u16* Abuf = alloc((size_t)ROWS * HD * 2);

  // 1) RMSNorm + cast
  rmsnorm_bf16_kernel<<<ROWS, 256, 0, stream>>>(tokens, norm_w, xh);

  // 2) weight conversions
  {
    int n1 = HD * HD, n2 = KVD * HD;
    cvt_f32_bf16_kernel<<<(n1 + 255) / 256, 256, 0, stream>>>(Wq, Wq_h, n1);
    cvt_f32_bf16_kernel<<<(n2 + 255) / 256, 256, 0, stream>>>(Wk, Wk_h, n2);
    cvt_f32_bf16_kernel<<<(n2 + 255) / 256, 256, 0, stream>>>(Wv, Wv_h, n2);
    cvt_f32_bf16_kernel<<<(n1 + 255) / 256, 256, 0, stream>>>(Wo, Wo_h, n1);
  }

  // 3) projections: Q/K/V (bf16 out)
  gemm_bf16_wmma_kernel<<<dim3(HD / BN, ROWS / BM), 256, 0, stream>>>(
      xh, Wq_h, bq, Qbuf, ROWS, HD, HD, 0);
  gemm_bf16_wmma_kernel<<<dim3(KVD / BN, ROWS / BM), 256, 0, stream>>>(
      xh, Wk_h, bk, Kbuf, ROWS, KVD, HD, 0);
  gemm_bf16_wmma_kernel<<<dim3(KVD / BN, ROWS / BM), 256, 0, stream>>>(
      xh, Wv_h, bv, Vbuf, ROWS, KVD, HD, 0);

  // 4) attention
  attn_wmma_kernel<<<Bb * NH * (Nn / 128), 256, 0, stream>>>(Qbuf, Kbuf, Vbuf, Abuf);

  // 5) output projection (f32 out)
  gemm_bf16_wmma_kernel<<<dim3(HD / BN, ROWS / BM), 256, 0, stream>>>(
      Abuf, Wo_h, bo, out, ROWS, HD, HD, 1);
}
